// ProposalNetwork_87840671138038
// MI455X (gfx1250) — compile-verified, matrix operand order checked by reading, not run on hardware
//
#include <hip/hip_runtime.h>
#include <hip/hip_bf16.h>

typedef __attribute__((ext_vector_type(16))) __bf16 v16bf;
typedef __attribute__((ext_vector_type(8)))  float  v8f;

namespace {

constexpr int LSEQ = 64, DMODEL = 256, NLAYER = 2, DFF = 512;
constexpr int KPROP = 128, VSZ = 110;

// bf16 weight workspace layout (element offsets)
constexpr size_t WS_PAIRW = 0;                                          // 256 x 512
constexpr size_t WS_WQKV  = WS_PAIRW + (size_t)DMODEL * (2 * DMODEL);   // NL x 768 x 256
constexpr size_t WS_WO    = WS_WQKV  + (size_t)NLAYER * 3 * DMODEL * DMODEL;
constexpr size_t WS_W1    = WS_WO    + (size_t)NLAYER * DMODEL * DMODEL;
constexpr size_t WS_W2    = WS_W1    + (size_t)NLAYER * DFF * DMODEL;
constexpr size_t WS_TOT   = WS_W2    + (size_t)NLAYER * DMODEL * DFF;   // 1,179,648 bf16

// LDS layout (byte offsets); total 289 KB -- legal on CDNA5 (320 KB/WGP)
constexpr int S_X  = 0;        // f32  64x256 residual stream
constexpr int S_A  = 65536;    // bf16 64x256 LN out / attn probs (4 x 64x64)
constexpr int S_B  = 98304;    // bf16 64x256 attn output
constexpr int S_Q  = 131072;   // bf16 64x256 ; with S_K also emb(64x512)/ffn hidden(64x512)
constexpr int S_K  = 163840;   // bf16 64x256
constexpr int S_VT = 196608;   // bf16 256x64 (V transposed, head-major)
constexpr int S_S  = 229376;   // f32  4x64x64 attention scores
constexpr int S_MU = 294912;
constexpr int S_RS = 295168;
constexpr int S_LG = 295424;
constexpr int S_PB = 295680;
constexpr int SMEM_BYTES = 295936;

__device__ __forceinline__ v16bf frag_ld(const __bf16* p0, int stride, int lane) {
  // 16-bit WMMA operand striping: lanes 0-15 rows r, K 0..7 & 16..23 ; lanes 16-31 K +8
  const __bf16* p = p0 + (lane & 15) * stride + ((lane >> 4) << 3);
  v16bf a;
  ((uint4*)&a)[0] = *(const uint4*)(p);
  ((uint4*)&a)[1] = *(const uint4*)(p + 16);
  return a;
}

#define WMMA_BF16(A_, B_, C_) \
  __builtin_amdgcn_wmma_f32_16x16x32_bf16(false, (A_), false, (B_), (short)0, (C_), false, false)

// Two 64x16 output strips sharing the A operand:
//   D0[0:64,0:16] = A * B0^T ; D1[0:64,0:16] = A * B1^T
// 4 A-frags + 2 B-frags feed 8 independent WMMAs per k-step: each ds_load
// feeds two WMMAs and the 8 independent chains hide LDS latency.
template <int KD, typename F>
__device__ __forceinline__ void mm_strip2(const __bf16* A, int As,
                                          const __bf16* B0, const __bf16* B1, int Bs,
                                          int lane, F&& epi) {
  __builtin_prefetch(B0, 0, 3);
  __builtin_prefetch(B1, 0, 3);
  v8f c00 = {}, c01 = {}, c02 = {}, c03 = {};
  v8f c10 = {}, c11 = {}, c12 = {}, c13 = {};
#pragma unroll
  for (int k0 = 0; k0 < KD; k0 += 32) {
    v16bf a0 = frag_ld(A + (size_t)0 * 16 * As + k0, As, lane);
    v16bf a1 = frag_ld(A + (size_t)1 * 16 * As + k0, As, lane);
    v16bf a2 = frag_ld(A + (size_t)2 * 16 * As + k0, As, lane);
    v16bf a3 = frag_ld(A + (size_t)3 * 16 * As + k0, As, lane);
    v16bf b0 = frag_ld(B0 + k0, Bs, lane);
    v16bf b1 = frag_ld(B1 + k0, Bs, lane);
    c00 = WMMA_BF16(a0, b0, c00);
    c01 = WMMA_BF16(a1, b0, c01);
    c02 = WMMA_BF16(a2, b0, c02);
    c03 = WMMA_BF16(a3, b0, c03);
    c10 = WMMA_BF16(a0, b1, c10);
    c11 = WMMA_BF16(a1, b1, c11);
    c12 = WMMA_BF16(a2, b1, c12);
    c13 = WMMA_BF16(a3, b1, c13);
  }
  epi(0, 0, c00); epi(0, 16, c01); epi(0, 32, c02); epi(0, 48, c03);
  epi(1, 0, c10); epi(1, 16, c11); epi(1, 32, c12); epi(1, 48, c13);
}

__device__ __forceinline__ void layer_norm(const float* __restrict__ x,
                                           const float* __restrict__ w,
                                           const float* __restrict__ bb,
                                           __bf16* __restrict__ o,
                                           float* s_mu, float* s_rs, int tid) {
  if (tid < 64) {
    const float* row = x + tid * DMODEL;
    float s = 0.f, ss = 0.f;
    for (int i = 0; i < DMODEL; ++i) { float v = row[i]; s += v; ss += v * v; }
    float mu = s * (1.f / 256.f);
    float var = ss * (1.f / 256.f) - mu * mu;
    s_mu[tid] = mu;
    s_rs[tid] = rsqrtf(var + 1e-5f);
  }
  __syncthreads();
  const int r = tid >> 2, c0 = (tid & 3) * 64;
  const float mu = s_mu[r], rs = s_rs[r];
  const float* row = x + r * DMODEL;
  __bf16* orow = o + r * DMODEL;
  for (int i = 0; i < 64; ++i) {
    int c = c0 + i;
    orow[c] = (__bf16)((row[c] - mu) * rs * w[c] + bb[c]);
  }
  __syncthreads();
}

__device__ __forceinline__ float urand(unsigned x) {
  x ^= 2747636419u; x *= 2654435769u;
  x ^= x >> 16;     x *= 2654435769u;
  x ^= x >> 16;     x *= 2654435769u;
  return (x >> 8) * (1.0f / 16777216.0f);
}

} // namespace

// fp32 weights -> bf16 workspace
__global__ void k_cvt_weights(const float* __restrict__ pw, const float* __restrict__ wqkv,
                              const float* __restrict__ wo, const float* __restrict__ w1,
                              const float* __restrict__ w2, __bf16* __restrict__ ws) {
  const size_t N0 = WS_WQKV - WS_PAIRW, N1 = WS_WO - WS_WQKV, N2 = WS_W1 - WS_WO,
               N3 = WS_W2 - WS_W1;
  for (size_t idx = blockIdx.x * (size_t)blockDim.x + threadIdx.x; idx < WS_TOT;
       idx += (size_t)gridDim.x * blockDim.x) {
    size_t o = idx;
    float v;
    if (o < N0) v = pw[o];
    else if ((o -= N0) < N1) v = wqkv[o];
    else if ((o -= N1) < N2) v = wo[o];
    else if ((o -= N2) < N3) v = w1[o];
    else v = w2[o - N3];
    ws[idx] = (__bf16)v;
  }
}

__global__ __launch_bounds__(256, 1) void k_fused(
    const int* __restrict__ genotypes,
    const float* __restrict__ allele_emb, const float* __restrict__ locus_emb,
    const float* __restrict__ pair_bv,
    const float* __restrict__ ln1_w, const float* __restrict__ ln1_b,
    const float* __restrict__ bqkv,  const float* __restrict__ bo,
    const float* __restrict__ ln2_w, const float* __restrict__ ln2_b,
    const float* __restrict__ b1,    const float* __restrict__ b2,
    const float* __restrict__ phase_W, const float* __restrict__ phase_b,
    const __bf16* __restrict__ ws,
    float* __restrict__ out_cand, float* __restrict__ out_logq,
    float* __restrict__ out_logits) {
  extern __shared__ char smem[];
  float*  s_x  = (float*) (smem + S_X);
  __bf16* s_a  = (__bf16*)(smem + S_A);
  __bf16* s_b  = (__bf16*)(smem + S_B);
  __bf16* s_q  = (__bf16*)(smem + S_Q);
  __bf16* s_k  = (__bf16*)(smem + S_K);
  __bf16* s_vt = (__bf16*)(smem + S_VT);
  float*  s_s  = (float*) (smem + S_S);
  float*  s_mu = (float*) (smem + S_MU);
  float*  s_rs = (float*) (smem + S_RS);
  float*  s_lg = (float*) (smem + S_LG);
  float*  s_pb = (float*) (smem + S_PB);
  __bf16* s_emb = s_q;  // 64 x 512 spans S_Q..S_K
  __bf16* s_h   = s_q;  // ffn hidden 64 x 512 reuses same span

  const int b    = blockIdx.x;
  const int tid  = threadIdx.x;
  const int lane = tid & 31;
  const int wave = tid >> 5;
  const int* g   = genotypes + (size_t)b * (2 * LSEQ);

  // ---- embedding gather: concat(emb[a0], emb[a1]) as bf16 (64 x 512) ----
  for (int e = tid; e < LSEQ * 512; e += 256) {
    int l = e >> 9, c = e & 511;
    int al = g[2 * l + (c >> 8)];
    al = al < 0 ? 0 : (al > VSZ - 1 ? VSZ - 1 : al);
    s_emb[e] = (__bf16)allele_emb[(size_t)al * DMODEL + (c & 255)];
  }
  __syncthreads();

  // ---- x = emb @ pairW^T + pair_b + locus_emb  (strips wave, wave+8) ----
  {
    const __bf16* Wp = ws + WS_PAIRW;
    int tn0 = wave, tn1 = wave + 8;
    mm_strip2<512>(s_emb, 512, Wp + (size_t)tn0 * 16 * 512, Wp + (size_t)tn1 * 16 * 512, 512, lane,
      [&](int s, int m0, v8f acc) {
        int n = (s ? tn1 : tn0) * 16 + (lane & 15), mb = m0 + ((lane >> 4) << 3);
        float bias = pair_bv[n];
        for (int i = 0; i < 8; ++i) {
          int m = mb + i;
          s_x[m * DMODEL + n] = acc[i] + bias + locus_emb[m * DMODEL + n];
        }
      });
  }
  __syncthreads();

  // ---- transformer layers ----
  for (int l = 0; l < NLAYER; ++l) {
    const __bf16* Wq  = ws + WS_WQKV + (size_t)l * 3 * DMODEL * DMODEL;
    const __bf16* Wol = ws + WS_WO   + (size_t)l * DMODEL * DMODEL;
    const __bf16* W1l = ws + WS_W1   + (size_t)l * DFF * DMODEL;
    const __bf16* W2l = ws + WS_W2   + (size_t)l * DMODEL * DFF;
    const float* bq  = bqkv + l * 3 * DMODEL;
    const float* bol = bo + l * DMODEL;
    const float* b1l = b1 + l * DFF;
    const float* b2l = b2 + l * DMODEL;

    layer_norm(s_x, ln1_w + l * DMODEL, ln1_b + l * DMODEL, s_a, s_mu, s_rs, tid);

    // QKV (store V transposed, head-major); strip pairs (t, t+8)
    for (int t = wave; t < 48; t += 16) {
      int tn0 = t, tn1 = t + 8;
      mm_strip2<256>(s_a, 256, Wq + (size_t)tn0 * 16 * 256, Wq + (size_t)tn1 * 16 * 256, 256, lane,
        [&](int s, int m0, v8f acc) {
          int n = (s ? tn1 : tn0) * 16 + (lane & 15), mb = m0 + ((lane >> 4) << 3);
          float bias = bq[n];
          for (int i = 0; i < 8; ++i) {
            int m = mb + i;
            float v = acc[i] + bias;
            if (n < 256)      s_q[m * 256 + n] = (__bf16)v;
            else if (n < 512) s_k[m * 256 + (n - 256)] = (__bf16)v;
            else              s_vt[(n - 512) * 64 + m] = (__bf16)v;
          }
        });
    }
    __syncthreads();

    // scores S = Q K^T: wave -> head h = wave>>1, strips tn0/tn0+1 in that head
    {
      int h = wave >> 1, tn0 = (wave & 1) * 2, tn1 = tn0 + 1;
      float* sh = s_s + h * 4096;
      const __bf16* Kb = s_k + h * 64;
      mm_strip2<64>(s_q + h * 64, 256,
                    Kb + (size_t)tn0 * 16 * 256, Kb + (size_t)tn1 * 16 * 256, 256, lane,
        [&](int s, int m0, v8f acc) {
          int n = (s ? tn1 : tn0) * 16 + (lane & 15), mb = m0 + ((lane >> 4) << 3);
          for (int i = 0; i < 8; ++i) sh[(mb + i) * 64 + n] = acc[i];
        });
    }
    __syncthreads();

    // softmax: one thread per (head,row); probs -> s_a (bf16, head-major)
    {
      int h = tid >> 6, r = tid & 63;
      float* srow = s_s + h * 4096 + r * 64;
      float mx = -1e30f;
      for (int m = 0; m < 64; ++m) mx = fmaxf(mx, srow[m] * 0.125f);
      float sum = 0.f;
      for (int m = 0; m < 64; ++m) {
        float e = __expf(srow[m] * 0.125f - mx);
        srow[m] = e; sum += e;
      }
      float inv = 1.f / sum;
      __bf16* prow = s_a + h * 4096 + r * 64;
      for (int m = 0; m < 64; ++m) prow[m] = (__bf16)(srow[m] * inv);
    }
    __syncthreads();

    // O = P V per head (B operand = rows of V^T); same head/strip mapping
    {
      int h = wave >> 1, tn0 = (wave & 1) * 2, tn1 = tn0 + 1;
      const __bf16* Vb = s_vt + h * 4096;
      mm_strip2<64>(s_a + h * 4096, 64,
                    Vb + (size_t)tn0 * 16 * 64, Vb + (size_t)tn1 * 16 * 64, 64, lane,
        [&](int s, int m0, v8f acc) {
          int n = (s ? tn1 : tn0) * 16 + (lane & 15), mb = m0 + ((lane >> 4) << 3);
          for (int i = 0; i < 8; ++i)
            s_b[(mb + i) * 256 + h * 64 + n] = (__bf16)acc[i];
        });
    }
    __syncthreads();

    // x += O Wo^T + bo
    {
      int tn0 = wave, tn1 = wave + 8;
      mm_strip2<256>(s_b, 256, Wol + (size_t)tn0 * 16 * 256, Wol + (size_t)tn1 * 16 * 256, 256, lane,
        [&](int s, int m0, v8f acc) {
          int n = (s ? tn1 : tn0) * 16 + (lane & 15), mb = m0 + ((lane >> 4) << 3);
          float bias = bol[n];
          for (int i = 0; i < 8; ++i) s_x[(mb + i) * 256 + n] += acc[i] + bias;
        });
    }
    __syncthreads();

    layer_norm(s_x, ln2_w + l * DMODEL, ln2_b + l * DMODEL, s_a, s_mu, s_rs, tid);

    // H = relu(A W1^T + b1) (bf16, 64x512); strip pairs (t, t+8)
    for (int t = wave; t < 32; t += 16) {
      int tn0 = t, tn1 = t + 8;
      mm_strip2<256>(s_a, 256, W1l + (size_t)tn0 * 16 * 256, W1l + (size_t)tn1 * 16 * 256, 256, lane,
        [&](int s, int m0, v8f acc) {
          int n = (s ? tn1 : tn0) * 16 + (lane & 15), mb = m0 + ((lane >> 4) << 3);
          float bias = b1l[n];
          for (int i = 0; i < 8; ++i)
            s_h[(mb + i) * 512 + n] = (__bf16)fmaxf(acc[i] + bias, 0.f);
        });
    }
    __syncthreads();

    // x += H W2^T + b2
    {
      int tn0 = wave, tn1 = wave + 8;
      mm_strip2<512>(s_h, 512, W2l + (size_t)tn0 * 16 * 512, W2l + (size_t)tn1 * 16 * 512, 512, lane,
        [&](int s, int m0, v8f acc) {
          int n = (s ? tn1 : tn0) * 16 + (lane & 15), mb = m0 + ((lane >> 4) << 3);
          float bias = b2l[n];
          for (int i = 0; i < 8; ++i) s_x[(mb + i) * 256 + n] += acc[i] + bias;
        });
    }
    __syncthreads();
  }

  // ---- head: logits / probs ----
  if (tid < 64) {
    float acc = phase_b[0];
    const float* xr = s_x + tid * DMODEL;
    for (int c = 0; c < DMODEL; ++c) acc += xr[c] * phase_W[c];
    s_lg[tid] = acc;
    s_pb[tid] = 1.f / (1.f + __expf(-acc));
    out_logits[(size_t)b * LSEQ + tid] = acc;
  }
  __syncthreads();

  // ---- proposals: candidates + log_q ----
  for (int kk = wave; kk < KPROP; kk += 8) {
    float part = 0.f;
    for (int half = 0; half < 2; ++half) {
      int l = lane + 32 * half;
      float p = s_pb[l];
      int ph;
      if (kk == 0)        ph = p > 0.5f;
      else if (kk <= 64)  ph = (int)(p > 0.5f) ^ (int)(l == kk - 1);
      else ph = urand(((unsigned)b * 16807u + (unsigned)kk * 2246822519u +
                       (unsigned)l * 3266489917u) ^ 42u) < p;
      float pf = (float)ph;
      part += pf * __logf(p + 1e-8f) + (1.f - pf) * __logf(1.f - p + 1e-8f);
      int a1 = g[2 * l], a2 = g[2 * l + 1];
      size_t o = (((size_t)b * KPROP + kk) * LSEQ + l) * 2;
      out_cand[o]     = (float)(ph ? a1 : a2);
      out_cand[o + 1] = (float)(ph ? a2 : a1);
    }
    for (int off = 16; off; off >>= 1) part += __shfl_xor(part, off, 32);
    if (lane == 0) out_logq[(size_t)b * KPROP + kk] = part;
  }
}

extern "C" void kernel_launch(void* const* d_in, const int* in_sizes, int n_in,
                              void* d_out, int out_size, void* d_ws, size_t ws_size,
                              hipStream_t stream) {
  const int*   genotypes  = (const int*)  d_in[0];
  const float* allele_emb = (const float*)d_in[1];
  const float* locus_emb  = (const float*)d_in[2];
  const float* pair_W     = (const float*)d_in[3];
  const float* pair_b     = (const float*)d_in[4];
  const float* ln1_w      = (const float*)d_in[5];
  const float* ln1_b      = (const float*)d_in[6];
  const float* Wqkv       = (const float*)d_in[7];
  const float* bqkv       = (const float*)d_in[8];
  const float* Wo         = (const float*)d_in[9];
  const float* bo         = (const float*)d_in[10];
  const float* ln2_w      = (const float*)d_in[11];
  const float* ln2_b      = (const float*)d_in[12];
  const float* W1         = (const float*)d_in[13];
  const float* b1         = (const float*)d_in[14];
  const float* W2         = (const float*)d_in[15];
  const float* b2         = (const float*)d_in[16];
  const float* phase_W    = (const float*)d_in[17];
  const float* phase_b    = (const float*)d_in[18];

  __bf16* ws = (__bf16*)d_ws;
  float*  out = (float*)d_out;
  const size_t CAND_N = (size_t)4096 * KPROP * LSEQ * 2;
  const size_t LOGQ_N = (size_t)4096 * KPROP;
  float* out_cand   = out;
  float* out_logq   = out + CAND_N;
  float* out_logits = out + CAND_N + LOGQ_N;

  k_cvt_weights<<<1024, 256, 0, stream>>>(pair_W, Wqkv, Wo, W1, W2, ws);
  k_fused<<<4096, 256, SMEM_BYTES, stream>>>(
      genotypes, allele_emb, locus_emb, pair_b, ln1_w, ln1_b, bqkv, bo,
      ln2_w, ln2_b, b1, b2, phase_W, phase_b, ws,
      out_cand, out_logq, out_logits);
}
